// MelPerformer_29326036697245
// MI455X (gfx1250) — matrix operand order repaired and checked by reading, not compiled, not run on hardware
//
#include <hip/hip_runtime.h>

#define DEV_INLINE __device__ __forceinline__

typedef __attribute__((ext_vector_type(16))) _Float16 v16h;
typedef __attribute__((ext_vector_type(8)))  _Float16 v8h;
typedef __attribute__((ext_vector_type(8)))  float    v8f;

constexpr int B_    = 8;
constexpr int N_    = 4096;
constexpr int NMELS = 64;
constexpr int DIM   = 256;
constexpr int LAYERS= 4;
constexpr int HEADS = 4;
constexpr int DHEAD = 64;
constexpr int NBF   = 256;   // random features
constexpr int G2    = 2048;  // GLU proj width
constexpr int FFH   = 1024;  // FF hidden
constexpr int ROWS  = B_ * N_;          // 32768
constexpr int BH    = B_ * HEADS;       // 32

// ---------------------------------------------------------------------------
// WMMA helpers (wave32, v_wmma_f32_16x16x32_f16)
// A fragment: lane&15 = M row; lanes<16 hold K {0..7,16..23}, lanes>=16 hold
// K {8..15,24..31} (two contiguous 16B LDS reads).
// B fragment: lane&15 = N col; lanes<16 hold K {0..7,16..23}, lanes>=16 hold
// K {8..15,24..31} (LDS staged N-major, two contiguous 16B reads).
// C/D: vgpr i holds M = i + 8*(lane>=16), N = lane&15.
// ---------------------------------------------------------------------------
DEV_INLINE v8f wmma16(v16h a, v16h b, v8f c) {
  return __builtin_amdgcn_wmma_f32_16x16x32_f16(false, a, false, b, (short)0, c,
                                                false, false);
}

DEV_INLINE v16h ldsA_frag(const _Float16* As, int r0, int kbase, int ldk, int lane) {
  const _Float16* p = As + (size_t)(r0 + (lane & 15)) * ldk + kbase + ((lane >> 4) << 3);
  v8h lo = *(const v8h*)(p);
  v8h hi = *(const v8h*)(p + 16);
  v16h r;
#pragma unroll
  for (int i = 0; i < 8; ++i) { r[i] = lo[i]; r[i + 8] = hi[i]; }
  return r;
}

DEV_INLINE v16h ldsB_frag(const _Float16* Bt, int c0, int kbase, int ldk, int lane) {
  const _Float16* p = Bt + (size_t)(c0 + (lane & 15)) * ldk + kbase + ((lane >> 4) << 3);
  v8h lo = *(const v8h*)(p);
  v8h hi = *(const v8h*)(p + 16);
  v16h r;
#pragma unroll
  for (int i = 0; i < 8; ++i) { r[i] = lo[i]; r[i + 8] = hi[i]; }
  return r;
}

// Stage a 64-row x 32-K tile, A kept row-major in LDS (dst stride 32).
// 128 threads: thread t covers row t/2, 16-half segment (t&1)*16.
DEV_INLINE void stage_direct(const _Float16* src0, int ld, _Float16* dst, int t) {
  int r = t >> 1, cs = (t & 1) * 16;
  const _Float16* src = src0 + (size_t)r * ld + cs;
  *(v8h*)(dst + r * 32 + cs)     = *(const v8h*)(src);
  *(v8h*)(dst + r * 32 + cs + 8) = *(const v8h*)(src + 8);
}

// Stage a 32-K x 64-col tile transposed into LDS: dst[c][k] (stride 32).
// 128 threads: thread t covers k = t/4, 16-col segment (t&3)*16.
DEV_INLINE void stage_transpose(const _Float16* src0, int ld, _Float16* dst, int t) {
  int k = t >> 2, cs = (t & 3) * 16;
  const _Float16* src = src0 + (size_t)k * ld + cs;
  v8h b0 = *(const v8h*)(src);
  v8h b1 = *(const v8h*)(src + 8);
#pragma unroll
  for (int j = 0; j < 8; ++j) {
    dst[(cs + j) * 32 + k]     = b0[j];
    dst[(cs + 8 + j) * 32 + k] = b1[j];
  }
}

DEV_INLINE unsigned fenc(float f) {
  unsigned u = __float_as_uint(f);
  return (u & 0x80000000u) ? ~u : (u | 0x80000000u);
}
DEV_INLINE float fdec(unsigned u) {
  unsigned b = (u & 0x80000000u) ? (u & 0x7FFFFFFFu) : ~u;
  return __uint_as_float(b);
}

// ---------------------------------------------------------------------------
// elementwise converts
// ---------------------------------------------------------------------------
__global__ __launch_bounds__(256) void cvt_kernel(const float* __restrict__ s,
                                                  _Float16* __restrict__ d, int n) {
  int i = blockIdx.x * 256 + threadIdx.x;
  if (i < n) d[i] = (_Float16)s[i];
}

// projT[d][m] = proj[m][d] * 64^-0.25   (fold data_normalizer into B matrix)
__global__ __launch_bounds__(256) void projT_kernel(const float* __restrict__ proj,
                                                    _Float16* __restrict__ projT) {
  int i = blockIdx.x * 256 + threadIdx.x;       // 64*256
  int d = i >> 8, m = i & 255;
  projT[i] = (_Float16)(proj[m * DHEAD + d] * 0.35355339059327373f);
}

// ---------------------------------------------------------------------------
// LayerNorm: one row per block, f32 in -> f16 out
// ---------------------------------------------------------------------------
__global__ __launch_bounds__(256) void ln_kernel(const float* __restrict__ h,
                                                 const float* __restrict__ g,
                                                 const float* __restrict__ b,
                                                 _Float16* __restrict__ y) {
  int row = blockIdx.x, t = threadIdx.x;
  float v = h[(size_t)row * DIM + t];
  __shared__ float s1[256], s2[256];
  s1[t] = v; s2[t] = v * v;
  __syncthreads();
  for (int s = 128; s > 0; s >>= 1) {
    if (t < s) { s1[t] += s1[t + s]; s2[t] += s2[t + s]; }
    __syncthreads();
  }
  float mu = s1[0] * (1.0f / DIM);
  float var = s2[0] * (1.0f / DIM) - mu * mu;
  float r = rsqrtf(var + 1e-5f);
  y[(size_t)row * DIM + t] = (_Float16)((v - mu) * r * g[t] + b[t]);
}

// ---------------------------------------------------------------------------
// Generic GEMM: out = A[M,K](f16) @ B[K,N](f16) + bias
// mode 0: write f16 -> outH ; mode 1: outF = residual + v ; mode 2: outF = v
// block = 128 threads (4 waves), tile 64x64, K-step 32, double-buffered LDS
// ---------------------------------------------------------------------------
__global__ __launch_bounds__(128)
void gemm_kernel(const _Float16* __restrict__ A, const _Float16* __restrict__ Bw,
                 const float* __restrict__ bias, int M, int N, int K,
                 float* __restrict__ outF, _Float16* __restrict__ outH,
                 const float* __restrict__ residual, int mode) {
  __shared__ _Float16 As[2][64 * 32];
  __shared__ _Float16 Bt[2][64 * 32];
  const int t = threadIdx.x, lane = t & 31, wave = t >> 5;
  const int rowBase = blockIdx.y * 64, colBase = blockIdx.x * 64;
  const _Float16* Abase = A + (size_t)rowBase * K;
  const _Float16* Bbase = Bw + colBase;
  v8f acc[4] = {};

  stage_direct(Abase, K, As[0], t);
  stage_transpose(Bbase, N, Bt[0], t);
  __syncthreads();

  const int nk = K >> 5;
  for (int s = 0; s < nk; ++s) {
    const int p = s & 1;
    if (s + 1 < nk) {
      stage_direct(Abase + (s + 1) * 32, K, As[p ^ 1], t);
      stage_transpose(Bbase + (size_t)(s + 1) * 32 * N, N, Bt[p ^ 1], t);
      if (s + 2 < nk) {  // pull K+2 tile toward L2/WGP$ (global_prefetch_b8)
        __builtin_prefetch(Abase + (size_t)(t >> 1) * K + (s + 2) * 32, 0, 1);
        __builtin_prefetch(Bbase + (size_t)((s + 2) * 32 + (t >> 2)) * N, 0, 1);
      }
    }
    v16h a = ldsA_frag(As[p], wave * 16, 0, 32, lane);
#pragma unroll
    for (int ct = 0; ct < 4; ++ct) {
      v16h b = ldsB_frag(Bt[p], ct * 16, 0, 32, lane);
      acc[ct] = wmma16(a, b, acc[ct]);
    }
    __syncthreads();
  }

  const int rOff = (lane >> 4) << 3, c16 = lane & 15;
#pragma unroll
  for (int ct = 0; ct < 4; ++ct) {
    int col = colBase + ct * 16 + c16;
    float bv = bias ? bias[col] : 0.0f;
#pragma unroll
    for (int i = 0; i < 8; ++i) {
      int row = rowBase + wave * 16 + rOff + i;
      size_t off = (size_t)row * N + col;
      float v = acc[ct][i] + bv;
      if (mode == 0)      outH[off] = (_Float16)v;
      else if (mode == 1) outF[off] = residual[off] + v;
      else                outF[off] = v;
    }
  }
}

// ---------------------------------------------------------------------------
// FAVOR+ feature kernel: dd = q_or_k @ projT (K=64, all 256 features in regs)
// isQuery: qp = ratio*(exp(dd - diag - rowmax)+eps)
// else:    write dd (f16) + diag, atomicMax global max of dd
// grid (N/64, B*H), block 128 (4 waves, 16 rows each)
// ---------------------------------------------------------------------------
__global__ __launch_bounds__(128)
void feature_kernel(const _Float16* __restrict__ qk, const _Float16* __restrict__ projT,
                    _Float16* __restrict__ outP, float* __restrict__ diagOut,
                    unsigned int* __restrict__ gmax, int isQuery) {
  __shared__ _Float16 As[64 * 64];
  __shared__ _Float16 Bt[64 * 64];
  __shared__ float diagS[64];
  __shared__ float redS[4];
  const int t = threadIdx.x, lane = t & 31, wave = t >> 5;
  const int bh = blockIdx.y, b = bh >> 2, h = bh & 3;
  const int n0 = blockIdx.x * 64;

  { // stage A: 64 rows x 64 head-dims
    int r = t >> 1, ds0 = (t & 1) * 32;
    const _Float16* src = qk + ((size_t)(b * N_ + n0 + r)) * DIM + h * DHEAD + ds0;
    _Float16* dst = As + r * 64 + ds0;
    *(v8h*)(dst)      = *(const v8h*)(src);
    *(v8h*)(dst + 8)  = *(const v8h*)(src + 8);
    *(v8h*)(dst + 16) = *(const v8h*)(src + 16);
    *(v8h*)(dst + 24) = *(const v8h*)(src + 24);
  }
  __syncthreads();
  if (t < 64) { // diag = 0.5*dn^2 * sum q^2  (dn^2 = 1/8)
    float s = 0.0f;
#pragma unroll 8
    for (int d = 0; d < 64; ++d) { float q = (float)As[t * 64 + d]; s += q * q; }
    diagS[t] = s * 0.0625f;
  }

  v8f acc[16] = {};
  for (int cc = 0; cc < 4; ++cc) {
    if (cc) __syncthreads();
    { // stage Bt chunk: Bt[c][d] = projT[d][cc*64+c]
      int d = t >> 1, cs = (t & 1) * 32;
      const _Float16* src = projT + (size_t)d * NBF + cc * 64 + cs;
      v8h p0 = *(const v8h*)(src);
      v8h p1 = *(const v8h*)(src + 8);
      v8h p2 = *(const v8h*)(src + 16);
      v8h p3 = *(const v8h*)(src + 24);
#pragma unroll
      for (int j = 0; j < 8; ++j) {
        Bt[(cs + j) * 64 + d]      = p0[j];
        Bt[(cs + 8 + j) * 64 + d]  = p1[j];
        Bt[(cs + 16 + j) * 64 + d] = p2[j];
        Bt[(cs + 24 + j) * 64 + d] = p3[j];
      }
    }
    __syncthreads();
#pragma unroll
    for (int ks = 0; ks < 64; ks += 32) {
      v16h a = ldsA_frag(As, wave * 16, ks, 64, lane);
#pragma unroll
      for (int ct = 0; ct < 4; ++ct) {
        v16h bb = ldsB_frag(Bt, ct * 16, ks, 64, lane);
        acc[cc * 4 + ct] = wmma16(a, bb, acc[cc * 4 + ct]);
      }
    }
  }
  __syncthreads();

  const float ratio = 0.0625f, eps = 1e-4f;
  const int rOff = (lane >> 4) << 3, c16 = lane & 15;

  if (isQuery) {
    float rmax[8];
#pragma unroll
    for (int i = 0; i < 8; ++i) {
      float m = acc[0][i];
#pragma unroll
      for (int ct = 1; ct < 16; ++ct) m = fmaxf(m, acc[ct][i]);
#pragma unroll
      for (int s = 1; s < 16; s <<= 1) m = fmaxf(m, __shfl_xor(m, s, 32));
      rmax[i] = m;
    }
#pragma unroll
    for (int ct = 0; ct < 16; ++ct) {
      int col = ct * 16 + c16;
#pragma unroll
      for (int i = 0; i < 8; ++i) {
        int lr = wave * 16 + rOff + i;
        float v = ratio * (expf(acc[ct][i] - diagS[lr] - rmax[i]) + eps);
        outP[((size_t)bh * N_ + n0 + lr) * NBF + col] = (_Float16)v;
      }
    }
  } else {
    float lm = -3.4e38f;
#pragma unroll
    for (int ct = 0; ct < 16; ++ct)
#pragma unroll
      for (int i = 0; i < 8; ++i) lm = fmaxf(lm, acc[ct][i]);
#pragma unroll
    for (int s = 1; s < 32; s <<= 1) lm = fmaxf(lm, __shfl_xor(lm, s, 32));
    if (lane == 0) redS[wave] = lm;
#pragma unroll
    for (int ct = 0; ct < 16; ++ct) {
      int col = ct * 16 + c16;
#pragma unroll
      for (int i = 0; i < 8; ++i) {
        int lr = wave * 16 + rOff + i;
        outP[((size_t)bh * N_ + n0 + lr) * NBF + col] = (_Float16)acc[ct][i];
      }
    }
    if (t < 64) diagOut[(size_t)bh * N_ + n0 + t] = diagS[t];
    __syncthreads();
    if (t == 0) {
      float bm = fmaxf(fmaxf(redS[0], redS[1]), fmaxf(redS[2], redS[3]));
      atomicMax(gmax, fenc(bm));
    }
  }
}

__global__ void init_kernel(unsigned int* g) { *g = 0x007FFFFFu; } // fenc(-inf)

// kp <- ratio*(exp(dd - diag - gmax)+eps)
__global__ __launch_bounds__(256)
void kexp_kernel(_Float16* __restrict__ kp, const float* __restrict__ diag,
                 const unsigned int* __restrict__ gmax) {
  size_t i = (size_t)blockIdx.x * 256 + threadIdx.x;
  float gm = fdec(*gmax);
  float dd = (float)kp[i];
  kp[i] = (_Float16)(0.0625f * (expf(dd - diag[i >> 8] - gm) + 1e-4f));
}

// kp_sum[bh][m] = sum_n kp[bh][n][m]
__global__ __launch_bounds__(256)
void kpsum_kernel(const _Float16* __restrict__ kp, float* __restrict__ ksum) {
  int bh = blockIdx.x, m = threadIdx.x;
  float s = 0.0f;
  for (int n = 0; n < N_; ++n) s += (float)kp[((size_t)bh * N_ + n) * NBF + m];
  ksum[bh * NBF + m] = s;
}

// d_inv[row] = 1 / (qp[row,:] . ksum[bh,:])
__global__ __launch_bounds__(256)
void dinv_kernel(const _Float16* __restrict__ qp, const float* __restrict__ ksum,
                 float* __restrict__ dinv) {
  int row = blockIdx.x, bh = row >> 12, m = threadIdx.x;
  __shared__ float red[256];
  red[m] = (float)qp[(size_t)row * NBF + m] * ksum[bh * NBF + m];
  __syncthreads();
  for (int s = 128; s > 0; s >>= 1) {
    if (m < s) red[m] += red[m + s];
    __syncthreads();
  }
  if (m == 0) dinv[row] = 1.0f / red[0];
}

// ctx[bh][m][d] = sum_n kp[bh][n][m] * v[b][n][h*64+d]  (K=4096, A=kp^T)
// grid (NBF/64, B*H), double-buffered
__global__ __launch_bounds__(128)
void ctx_kernel(const _Float16* __restrict__ kp, const _Float16* __restrict__ v,
                _Float16* __restrict__ ctx) {
  __shared__ _Float16 As[2][64 * 32];  // As[m][n]
  __shared__ _Float16 Bt[2][64 * 32];  // Bt[d][n]
  const int t = threadIdx.x, lane = t & 31, wave = t >> 5;
  const int bh = blockIdx.y, b = bh >> 2, h = bh & 3;
  const int m0 = blockIdx.x * 64;
  const _Float16* Abase = kp + (size_t)bh * N_ * NBF + m0;         // + n*NBF
  const _Float16* Bbase = v + (size_t)b * N_ * DIM + h * DHEAD;    // + n*DIM
  v8f acc[4] = {};

  stage_transpose(Abase, NBF, As[0], t);
  stage_transpose(Bbase, DIM, Bt[0], t);
  __syncthreads();

  const int nk = N_ >> 5;  // 128 steps
  for (int s = 0; s < nk; ++s) {
    const int p = s & 1;
    if (s + 1 < nk) {
      stage_transpose(Abase + (size_t)(s + 1) * 32 * NBF, NBF, As[p ^ 1], t);
      stage_transpose(Bbase + (size_t)(s + 1) * 32 * DIM, DIM, Bt[p ^ 1], t);
      if (s + 2 < nk) {
        __builtin_prefetch(Abase + (size_t)((s + 2) * 32 + (t >> 2)) * NBF, 0, 1);
        __builtin_prefetch(Bbase + (size_t)((s + 2) * 32 + (t >> 2)) * DIM, 0, 1);
      }
    }
    v16h a = ldsA_frag(As[p], wave * 16, 0, 32, lane);
#pragma unroll
    for (int ct = 0; ct < 4; ++ct) {
      v16h bb = ldsB_frag(Bt[p], ct * 16, 0, 32, lane);
      acc[ct] = wmma16(a, bb, acc[ct]);
    }
    __syncthreads();
  }

  const int rOff = (lane >> 4) << 3, c16 = lane & 15;
#pragma unroll
  for (int ct = 0; ct < 4; ++ct)
#pragma unroll
    for (int i = 0; i < 8; ++i) {
      int m = m0 + wave * 16 + rOff + i;
      ctx[((size_t)bh * NBF + m) * DHEAD + ct * 16 + c16] = (_Float16)acc[ct][i];
    }
}

// o[b][n][h*64+d] = d_inv[bh n] * sum_m qp[bh][n][m] * ctx[bh][m][d]
// grid (N/64, B*H), double-buffered
__global__ __launch_bounds__(128)
void o_kernel(const _Float16* __restrict__ qp, const _Float16* __restrict__ ctx,
              const float* __restrict__ dinv, _Float16* __restrict__ o16) {
  __shared__ _Float16 As[2][64 * 32];
  __shared__ _Float16 Bt[2][64 * 32];
  const int t = threadIdx.x, lane = t & 31, wave = t >> 5;
  const int bh = blockIdx.y, b = bh >> 2, h = bh & 3;
  const int n0 = blockIdx.x * 64;
  const _Float16* Abase = qp + ((size_t)bh * N_ + n0) * NBF;   // + k
  const _Float16* Bbase = ctx + (size_t)bh * NBF * DHEAD;      // + k*DHEAD
  v8f acc[4] = {};

  stage_direct(Abase, NBF, As[0], t);
  stage_transpose(Bbase, DHEAD, Bt[0], t);
  __syncthreads();

  const int nk = NBF >> 5;  // 8 steps
  for (int s = 0; s < nk; ++s) {
    const int p = s & 1;
    if (s + 1 < nk) {
      stage_direct(Abase + (s + 1) * 32, NBF, As[p ^ 1], t);
      stage_transpose(Bbase + (size_t)(s + 1) * 32 * DHEAD, DHEAD, Bt[p ^ 1], t);
    }
    v16h a = ldsA_frag(As[p], wave * 16, 0, 32, lane);
#pragma unroll
    for (int ct = 0; ct < 4; ++ct) {
      v16h bb = ldsB_frag(Bt[p], ct * 16, 0, 32, lane);
      acc[ct] = wmma16(a, bb, acc[ct]);
    }
    __syncthreads();
  }

  const int rOff = (lane >> 4) << 3, c16 = lane & 15;
#pragma unroll
  for (int ct = 0; ct < 4; ++ct)
#pragma unroll
    for (int i = 0; i < 8; ++i) {
      int n = n0 + wave * 16 + rOff + i;
      float di = dinv[(size_t)bh * N_ + n];
      o16[((size_t)(b * N_ + n)) * DIM + h * DHEAD + ct * 16 + c16] =
          (_Float16)(acc[ct][i] * di);
    }
}

// GLU FF: t[n][j] = gelu(y@w1[:,j]+b1[j]) * (y@w1[:,1024+j]+b1[1024+j])
// grid (FFH/64, ROWS/64), double-buffered
__global__ __launch_bounds__(128)
void glu_kernel(const _Float16* __restrict__ y, const _Float16* __restrict__ w1,
                const float* __restrict__ b1, _Float16* __restrict__ tOut) {
  __shared__ _Float16 As[2][64 * 32];
  __shared__ _Float16 BtA[2][64 * 32];
  __shared__ _Float16 BtG[2][64 * 32];
  const int t = threadIdx.x, lane = t & 31, wave = t >> 5;
  const int rowBase = blockIdx.y * 64, colBase = blockIdx.x * 64;
  const _Float16* Abase  = y + (size_t)rowBase * DIM;
  const _Float16* BbaseA = w1 + colBase;
  const _Float16* BbaseG = w1 + colBase + FFH;
  v8f accA[4] = {}, accG[4] = {};

  stage_direct(Abase, DIM, As[0], t);
  stage_transpose(BbaseA, G2, BtA[0], t);
  stage_transpose(BbaseG, G2, BtG[0], t);
  __syncthreads();

  const int nk = DIM >> 5;  // 8 steps
  for (int s = 0; s < nk; ++s) {
    const int p = s & 1;
    if (s + 1 < nk) {
      stage_direct(Abase + (s + 1) * 32, DIM, As[p ^ 1], t);
      stage_transpose(BbaseA + (size_t)(s + 1) * 32 * G2, G2, BtA[p ^ 1], t);
      stage_transpose(BbaseG + (size_t)(s + 1) * 32 * G2, G2, BtG[p ^ 1], t);
      if (s + 2 < nk) {
        __builtin_prefetch(BbaseA + (size_t)((s + 2) * 32 + (t >> 2)) * G2, 0, 1);
        __builtin_prefetch(BbaseG + (size_t)((s + 2) * 32 + (t >> 2)) * G2, 0, 1);
      }
    }
    v16h a = ldsA_frag(As[p], wave * 16, 0, 32, lane);
#pragma unroll
    for (int ct = 0; ct < 4; ++ct) {
      v16h bA = ldsB_frag(BtA[p], ct * 16, 0, 32, lane);
      accA[ct] = wmma16(a, bA, accA[ct]);
      v16h bG = ldsB_frag(BtG[p], ct * 16, 0, 32, lane);
      accG[ct] = wmma16(a, bG, accG[ct]);
    }
    __syncthreads();
  }

  const int rOff = (lane >> 4) << 3, c16 = lane & 15;
#pragma unroll
  for (int ct = 0; ct < 4; ++ct) {
    int col = colBase + ct * 16 + c16;
    float ba = b1[col], bg = b1[FFH + col];
#pragma unroll
    for (int i = 0; i < 8; ++i) {
      int row = rowBase + wave * 16 + rOff + i;
      float av = accA[ct][i] + ba;
      float gv = accG[ct][i] + bg;
      float ge = 0.5f * av * (1.0f + erff(av * 0.70710678118654752f));
      tOut[(size_t)row * FFH + col] = (_Float16)(ge * gv);
    }
  }
}

// mean over time + classifier: out[b] = mean_n(h[b,n,:]) . clf_w + clf_b
__global__ __launch_bounds__(256)
void clf_kernel(const float* __restrict__ h, const float* __restrict__ w,
                const float* __restrict__ bias, float* __restrict__ out) {
  int b = blockIdx.x, d = threadIdx.x;
  float s = 0.0f;
  for (int n = 0; n < N_; ++n) s += h[((size_t)(b * N_ + n)) * DIM + d];
  __shared__ float red[256];
  red[d] = s * (1.0f / N_) * w[d];
  __syncthreads();
  for (int st = 128; st > 0; st >>= 1) {
    if (d < st) red[d] += red[d + st];
    __syncthreads();
  }
  if (d == 0) out[b] = red[0] + bias[0];
}

// ---------------------------------------------------------------------------
extern "C" void kernel_launch(void* const* d_in, const int* in_sizes, int n_in,
                              void* d_out, int out_size, void* d_ws, size_t ws_size,
                              hipStream_t stream) {
  (void)in_sizes; (void)n_in; (void)out_size; (void)ws_size;
  const float* x      = (const float*)d_in[0];
  const float* proj_w = (const float*)d_in[1];
  const float* proj_b = (const float*)d_in[2];
  const float* ln1_g  = (const float*)d_in[3];
  const float* ln1_b  = (const float*)d_in[4];
  const float* wq     = (const float*)d_in[5];
  const float* bq     = (const float*)d_in[6];
  const float* wk     = (const float*)d_in[7];
  const float* bk     = (const float*)d_in[8];
  const float* wv     = (const float*)d_in[9];
  const float* bv     = (const float*)d_in[10];
  const float* wo     = (const float*)d_in[11];
  const float* bo     = (const float*)d_in[12];
  const float* ln2_g  = (const float*)d_in[13];
  const float* ln2_b  = (const float*)d_in[14];
  const float* w1     = (const float*)d_in[15];
  const float* b1     = (const float*)d_in[16];
  const float* w2     = (const float*)d_in[17];
  const float* b2     = (const float*)d_in[18];
  const float* clf_w  = (const float*)d_in[19];
  const float* clf_b  = (const float*)d_in[20];
  const float* proj_m = (const float*)d_in[21];
  float* out = (float*)d_out;

  char* ws = (char*)d_ws;
  size_t cur = 0;
  auto alloc = [&](size_t bytes) -> void* {
    void* p = ws + cur;
    cur += (bytes + 255) & ~(size_t)255;
    return p;
  };
  float*     h     = (float*)    alloc((size_t)ROWS * DIM * 4);
  _Float16*  y16   = (_Float16*) alloc((size_t)ROWS * DIM * 2);
  _Float16*  x16   = (_Float16*) alloc((size_t)ROWS * NMELS * 2);
  _Float16*  q16   = (_Float16*) alloc((size_t)ROWS * DIM * 2);
  _Float16*  k16   = (_Float16*) alloc((size_t)ROWS * DIM * 2);
  _Float16*  v16_  = (_Float16*) alloc((size_t)ROWS * DIM * 2);
  _Float16*  o16   = (_Float16*) alloc((size_t)ROWS * DIM * 2);
  _Float16*  qp    = (_Float16*) alloc((size_t)BH * N_ * NBF * 2);
  _Float16*  kp    = (_Float16*) alloc((size_t)BH * N_ * NBF * 2);  // reused as GLU hidden
  _Float16*  ctx16 = (_Float16*) alloc((size_t)BH * NBF * DHEAD * 2);
  float*     diagk = (float*)    alloc((size_t)BH * N_ * 4);
  float*     ksum  = (float*)    alloc((size_t)BH * NBF * 4);
  float*     dinv  = (float*)    alloc((size_t)BH * N_ * 4);
  unsigned*  gmax  = (unsigned*) alloc(256);
  _Float16*  pw16  = (_Float16*) alloc((size_t)NMELS * DIM * 2);
  _Float16*  pT16  = (_Float16*) alloc((size_t)DHEAD * NBF * 2);
  _Float16*  wq16  = (_Float16*) alloc((size_t)LAYERS * DIM * DIM * 2);
  _Float16*  wk16  = (_Float16*) alloc((size_t)LAYERS * DIM * DIM * 2);
  _Float16*  wv16  = (_Float16*) alloc((size_t)LAYERS * DIM * DIM * 2);
  _Float16*  wo16  = (_Float16*) alloc((size_t)LAYERS * DIM * DIM * 2);
  _Float16*  w116  = (_Float16*) alloc((size_t)LAYERS * DIM * G2 * 2);
  _Float16*  w216  = (_Float16*) alloc((size_t)LAYERS * FFH * DIM * 2);
  _Float16*  t16   = kp;  // GLU hidden aliases kp (dead by then)

  auto cvt = [&](const float* s, _Float16* d, int n) {
    cvt_kernel<<<dim3((n + 255) / 256), dim3(256), 0, stream>>>(s, d, n);
  };
  cvt(x, x16, ROWS * NMELS);
  cvt(proj_w, pw16, NMELS * DIM);
  cvt(wq, wq16, LAYERS * DIM * DIM);
  cvt(wk, wk16, LAYERS * DIM * DIM);
  cvt(wv, wv16, LAYERS * DIM * DIM);
  cvt(wo, wo16, LAYERS * DIM * DIM);
  cvt(w1, w116, LAYERS * DIM * G2);
  cvt(w2, w216, LAYERS * FFH * DIM);
  projT_kernel<<<dim3((DHEAD * NBF) / 256), dim3(256), 0, stream>>>(proj_m, pT16);

  auto gemm = [&](const _Float16* A, const _Float16* Bw, const float* bias,
                  int M, int N, int K, float* oF, _Float16* oH,
                  const float* res, int mode) {
    gemm_kernel<<<dim3(N / 64, M / 64), dim3(128), 0, stream>>>(
        A, Bw, bias, M, N, K, oF, oH, res, mode);
  };

  // h = x @ proj_w + proj_b
  gemm(x16, pw16, proj_b, ROWS, DIM, NMELS, h, nullptr, nullptr, 2);

  for (int l = 0; l < LAYERS; ++l) {
    // --- attention ---
    ln_kernel<<<dim3(ROWS), dim3(256), 0, stream>>>(h, ln1_g + l * DIM, ln1_b + l * DIM, y16);
    gemm(y16, wq16 + (size_t)l * DIM * DIM, bq + l * DIM, ROWS, DIM, DIM, nullptr, q16, nullptr, 0);
    gemm(y16, wk16 + (size_t)l * DIM * DIM, bk + l * DIM, ROWS, DIM, DIM, nullptr, k16, nullptr, 0);
    gemm(y16, wv16 + (size_t)l * DIM * DIM, bv + l * DIM, ROWS, DIM, DIM, nullptr, v16_, nullptr, 0);

    feature_kernel<<<dim3(N_ / 64, BH), dim3(128), 0, stream>>>(
        q16, pT16, qp, nullptr, nullptr, 1);
    init_kernel<<<dim3(1), dim3(1), 0, stream>>>(gmax);
    feature_kernel<<<dim3(N_ / 64, BH), dim3(128), 0, stream>>>(
        k16, pT16, kp, diagk, gmax, 0);
    kexp_kernel<<<dim3((unsigned)((size_t)BH * N_ * NBF / 256)), dim3(256), 0, stream>>>(
        kp, diagk, gmax);
    kpsum_kernel<<<dim3(BH), dim3(256), 0, stream>>>(kp, ksum);
    ctx_kernel<<<dim3(NBF / 64, BH), dim3(128), 0, stream>>>(kp, v16_, ctx16);
    dinv_kernel<<<dim3(BH * N_), dim3(256), 0, stream>>>(qp, ksum, dinv);
    o_kernel<<<dim3(N_ / 64, BH), dim3(128), 0, stream>>>(qp, ctx16, dinv, o16);
    gemm(o16, wo16 + (size_t)l * DIM * DIM, bo + l * DIM, ROWS, DIM, DIM, h, nullptr, h, 1);

    // --- GLU feed-forward ---
    ln_kernel<<<dim3(ROWS), dim3(256), 0, stream>>>(h, ln2_g + l * DIM, ln2_b + l * DIM, y16);
    glu_kernel<<<dim3(FFH / 64, ROWS / 64), dim3(128), 0, stream>>>(
        y16, w116 + (size_t)l * DIM * G2, b1 + (size_t)l * G2, t16);
    gemm(t16, w216 + (size_t)l * FFH * DIM, b2 + l * DIM, ROWS, DIM, FFH, h, nullptr, h, 1);
  }

  clf_kernel<<<dim3(B_), dim3(256), 0, stream>>>(h, clf_w, clf_b, out);
}
